// MultiHeadAttention_35227321762402
// MI455X (gfx1250) — matrix operand order repaired and checked by reading, not compiled, not run on hardware
//
#include <hip/hip_runtime.h>

// ---------------------------------------------------------------------------
// MHA for MI455X (gfx1250): all GEMM stages via v_wmma_f32_16x16x32_f16.
// Round 3: double-buffer fills now use CDNA5 async global->LDS copies
// (global_load_async_to_lds_b128 / ASYNCcnt) instead of global->VGPR->LDS.
// B=2, S=2048, HIDDEN=1024, HEADS=16, HEAD_DIM=64. Inputs f32, output f32.
// ---------------------------------------------------------------------------

#define HID 1024
#define NB  2
#define NS  2048
#define NH  16
#define ND  64

typedef _Float16 v16h __attribute__((ext_vector_type(16)));
typedef float    v8f  __attribute__((ext_vector_type(8)));

union Frag {
    v16h v;
    uint4 q[2];
    unsigned short u16[16];
};

// lane L <-> lane L^16 exchange (group-of-32 swizzle: and=0x1f, xor=0x10)
__device__ __forceinline__ float swzx16(float x) {
    return __int_as_float(__builtin_amdgcn_ds_swizzle(__float_as_int(x), 0x401F));
}
__device__ __forceinline__ int swzx16i(int x) {
    return __builtin_amdgcn_ds_swizzle(x, 0x401F);
}

__device__ __forceinline__ unsigned short f2h_bits(float x) {
    union { _Float16 h; unsigned short u; } c;
    c.h = (_Float16)x;
    return c.u;
}

__device__ __forceinline__ v8f wmma16(v16h a, v16h b, v8f c) {
    return __builtin_amdgcn_wmma_f32_16x16x32_f16(false, a, false, b, (short)0, c,
                                                  false, false);
}

// ---------------------------------------------------------------------------
// CDNA5 async global->LDS copy (16B per lane), tracked by ASYNCcnt.
// VDST VGPR holds the LDS byte offset (= low 32 bits of the flat LDS address,
// per the LDS aperture mapping); VADDR holds the 64-bit global address.
// ---------------------------------------------------------------------------
__device__ __forceinline__ void async_copy_b128(void* lds_dst, const void* gsrc) {
    unsigned loff = (unsigned)(size_t)lds_dst;
    unsigned long long ga = (unsigned long long)(size_t)gsrc;
    asm volatile("global_load_async_to_lds_b128 %0, %1, off"
                 :: "v"(loff), "v"(ga)
                 : "memory");
}
__device__ __forceinline__ void async_wait0() {
    asm volatile("s_wait_asynccnt 0" ::: "memory");
}

// A-operand fragment (M=16, K=32, fp16) from row-major [rows, ld]:
// lane L: row = row0 + (L&15); elems 0..15 <- k0 + 16*(L>>4) + (0..15)
__device__ __forceinline__ v16h load_a(const _Float16* base, int ld, int row0,
                                       int k0, int lane) {
    int r = row0 + (lane & 15);
    int k = k0 + ((lane >> 4) << 4);
    const _Float16* p = base + (size_t)r * ld + k;
    Frag f;
    f.q[0] = *(const uint4*)(p);
    f.q[1] = *(const uint4*)(p + 8);
    return f.v;
}

// B-operand fragment (K=32, N=16, fp16), NT use (source row n holds B[:,n]):
// lane L: col = col0 + (L&15); elems 0..15 <- k0 + 16*(L>>4) + (0..15)
__device__ __forceinline__ v16h load_b(const _Float16* base, int ld, int col0,
                                       int k0, int lane) {
    int c = col0 + (lane & 15);
    int k = k0 + ((lane >> 4) << 4);
    const _Float16* p = base + (size_t)c * ld + k;
    Frag f;
    f.q[0] = *(const uint4*)(p);
    f.q[1] = *(const uint4*)(p + 8);
    return f.v;
}

// ---------------------------------------------------------------------------
// fp32 -> fp16 conversion
// ---------------------------------------------------------------------------
__global__ void cvt_kernel(const float* __restrict__ src, _Float16* __restrict__ dst,
                           int n) {
    int i = blockIdx.x * blockDim.x + threadIdx.x;
    if (i < n) dst[i] = (_Float16)src[i];
}

// ---------------------------------------------------------------------------
// NT GEMM: D[m,n] = sum_k A[m,k] * W[n,k] + bias[n]
// Block = 8 waves: 128(m) x 64(n) tile; waves share the 64x32 weight tile
// through double-buffered LDS filled by async global->LDS copies.
// mode 0: fp16 [B,H,S,D]; mode 2: fp16 [B,H,D,S] (V^T); mode 3: fp32 [m,HID]
// ---------------------------------------------------------------------------
__global__ void gemm_nt_kernel(const _Float16* __restrict__ A,
                               const _Float16* __restrict__ W,
                               const float* __restrict__ bias,
                               _Float16* __restrict__ outh,
                               float* __restrict__ outf, int mode) {
    __shared__ _Float16 sW[2][64 * 32];   // [n_local][k_local], double buffered

    const int tid    = threadIdx.x;       // 0..255
    const int lane   = tid & 31;
    const int waveid = tid >> 5;          // 0..7
    const int mg = blockIdx.x >> 4;       // 32 m-groups of 128 rows
    const int ng = blockIdx.x & 15;       // 16 n-groups of 64 cols
    const int m0 = (mg * 8 + waveid) * 16;
    const int n0 = ng * 64;

    // cooperative fill: 64 rows x 32 halfs, 256 threads x 8 halfs (b128 each)
    const int frow = tid >> 2;            // 0..63  (n local)
    const int fcol = (tid & 3) * 8;       // 0,8,16,24 (k local)
    const _Float16* wsrc = W + (size_t)(n0 + frow) * HID + fcol;

    async_copy_b128(&sW[0][frow * 32 + fcol], wsrc);
    async_wait0();
    __syncthreads();

    v8f acc[4] = {};
    int cur = 0;
    for (int k = 0; k < HID; k += 32) {
        // read this step's fragments from LDS into VGPRs
        v16h b0 = load_b(&sW[cur][0], 32, 0,  0, lane);
        v16h b1 = load_b(&sW[cur][0], 32, 16, 0, lane);
        v16h b2 = load_b(&sW[cur][0], 32, 32, 0, lane);
        v16h b3 = load_b(&sW[cur][0], 32, 48, 0, lane);
        v16h a  = load_a(A, HID, m0, k, lane);

        // async-prefetch next weight tile into the other buffer
        if (k + 32 < HID)
            async_copy_b128(&sW[cur ^ 1][frow * 32 + fcol], wsrc + (k + 32));

        acc[0] = wmma16(a, b0, acc[0]);
        acc[1] = wmma16(a, b1, acc[1]);
        acc[2] = wmma16(a, b2, acc[2]);
        acc[3] = wmma16(a, b3, acc[3]);

        async_wait0();     // async fill LDS-visible before the barrier
        __syncthreads();
        cur ^= 1;
    }

    // C/D layout: lane L -> n = (L&15); VGPR j -> m = j + 8*(L>>4)
    const int qn   = lane & 15;
    const int half = lane >> 4;
#pragma unroll
    for (int t = 0; t < 4; ++t) {
        int n = n0 + t * 16 + qn;
        float bv = bias[n];
#pragma unroll
        for (int j = 0; j < 8; ++j) {
            int m = m0 + j + 8 * half;
            float v = acc[t][j] + bv;
            if (mode == 3) {
                outf[(size_t)m * HID + n] = v;
            } else {
                int b = m >> 11, s = m & (NS - 1);
                int h = n >> 6, d = n & (ND - 1);
                size_t idx = (mode == 2)
                                 ? ((((size_t)b * NH + h) * ND + d) * NS + s)
                                 : ((((size_t)b * NH + h) * NS + s) * ND + d);
                outh[idx] = (_Float16)v;
            }
        }
    }
}

// ---------------------------------------------------------------------------
// Flash attention. Block = 8 waves, all on the same (b,h), consecutive
// q-tiles; the 32-key K (32x64) and V^T (64x32) blocks are shared through
// double-buffered LDS filled by async global->LDS copies (8x less L2 traffic
// than per-wave loads). Computes S^T tiles so softmax reductions are
// intra-lane + one XOR-16 swizzle, and P^T feeds the PV WMMA's B operand
// fully in-register.
// ---------------------------------------------------------------------------
__global__ void attn_kernel(const _Float16* __restrict__ Qb,
                            const _Float16* __restrict__ Kb,
                            const _Float16* __restrict__ Vtb,
                            const float* __restrict__ mask,
                            _Float16* __restrict__ AOb) {
    __shared__ _Float16 sK[2][32 * 64];   // [key_local][d]
    __shared__ _Float16 sV[2][64 * 32];   // [d][key_local]

    const int tid    = threadIdx.x;
    const int lane   = tid & 31;
    const int waveid = tid >> 5;
    const int bh = blockIdx.x >> 4;              // 32 (b,h) pairs
    const int qg = blockIdx.x & 15;              // 16 q-groups of 8 tiles
    const int qt = qg * 8 + waveid;
    const int b = bh >> 4, h = bh & 15;

    const _Float16* Q  = Qb  + (size_t)bh * NS * ND;
    const _Float16* K  = Kb  + (size_t)bh * NS * ND;
    const _Float16* Vt = Vtb + (size_t)bh * ND * NS;
    const float*    mk = mask + (size_t)b * NS;

    // cooperative fill indices
    const int krow = tid >> 3, kcol = (tid & 7) * 8;   // K: 32 x 64
    const int vrow = tid >> 2, vcol = (tid & 3) * 8;   // V^T: 64 x 32

    // Q fragments (B operand), loaded once: d = 0..31 and 32..63
    v16h bq0 = load_b(Q, ND, qt * 16, 0, lane);
    v16h bq1 = load_b(Q, ND, qt * 16, 32, lane);

    // prologue async fill of buffer 0 (keys 0..31)
    async_copy_b128(&sK[0][krow * 64 + kcol], K + (size_t)krow * ND + kcol);
    async_copy_b128(&sV[0][vrow * 32 + vcol], Vt + (size_t)vrow * NS + vcol);
    async_wait0();
    __syncthreads();

    v8f acc[4] = {};            // out^T: acc[t] holds d = t*16 + (j+8*half), col q
    float mi = -1e30f, li = 0.0f;
    const int half = lane >> 4;
    const float sc = 0.125f;    // 1/sqrt(64)
    int cur = 0;

    for (int k0 = 0; k0 < NS; k0 += 32) {
        // read fragments for this key block from LDS
        v16h ak0 = load_a(&sK[cur][0], 64, 0,  0,  lane);
        v16h ak1 = load_a(&sK[cur][0], 64, 0,  32, lane);
        v16h ak2 = load_a(&sK[cur][0], 64, 16, 0,  lane);
        v16h ak3 = load_a(&sK[cur][0], 64, 16, 32, lane);
        v16h av0 = load_a(&sV[cur][0], 32, 0,  0, lane);
        v16h av1 = load_a(&sV[cur][0], 32, 16, 0, lane);
        v16h av2 = load_a(&sV[cur][0], 32, 32, 0, lane);
        v16h av3 = load_a(&sV[cur][0], 32, 48, 0, lane);

        // async-prefetch next key block into the other buffer
        if (k0 + 32 < NS) {
            async_copy_b128(&sK[cur ^ 1][krow * 64 + kcol],
                            K + (size_t)(k0 + 32 + krow) * ND + kcol);
            async_copy_b128(&sV[cur ^ 1][vrow * 32 + vcol],
                            Vt + (size_t)vrow * NS + (k0 + 32) + vcol);
        }

        // scores^T tiles: tile0 = keys k0..k0+15, tile1 = k0+16..k0+31
        v8f s0 = {}, s1 = {};
        s0 = wmma16(ak0, bq0, s0);
        s0 = wmma16(ak1, bq1, s0);
        s1 = wmma16(ak2, bq0, s1);
        s1 = wmma16(ak3, bq1, s1);

        // scale + additive mask; per-lane: 16 of the 32 keys for q = lane&15
        float e0[8], e1[8];
        float pmax = -1e30f;
#pragma unroll
        for (int j = 0; j < 8; ++j) {
            e0[j] = s0[j] * sc + mk[k0 + 8 * half + j];
            e1[j] = s1[j] * sc + mk[k0 + 16 + 8 * half + j];
            pmax = fmaxf(pmax, fmaxf(e0[j], e1[j]));
        }
        pmax = fmaxf(pmax, swzx16(pmax));    // combine lane halves (same q)
        float mnew  = fmaxf(mi, pmax);
        float alpha = __expf(mi - mnew);
        mi = mnew;

        float psum = 0.0f;
        unsigned up[8];                       // packed {p1,p0} fp16 pairs
#pragma unroll
        for (int j = 0; j < 8; ++j) {
            float p0 = __expf(e0[j] - mnew);
            float p1 = __expf(e1[j] - mnew);
            psum += p0 + p1;
            up[j] = (unsigned)f2h_bits(p0) | ((unsigned)f2h_bits(p1) << 16);
        }
        psum += swzx16(psum);
        li = li * alpha + psum;

        // P^T -> B operand: lane L (q=L&15) needs keys 16*(L>>4)+(0..15);
        // own lane has 8, lane^16 has the other 8.
        Frag pf;
#pragma unroll
        for (int j = 0; j < 8; ++j) {
            unsigned op = (unsigned)swzx16i((int)up[j]);
            unsigned short mine_lo = (unsigned short)(up[j] & 0xFFFFu);
            unsigned short mine_hi = (unsigned short)(up[j] >> 16);
            unsigned short oth_lo  = (unsigned short)(op & 0xFFFFu);
            unsigned short oth_hi  = (unsigned short)(op >> 16);
            pf.u16[j]     = half ? oth_hi : mine_lo;
            pf.u16[8 + j] = half ? mine_hi : oth_lo;
        }

        // rescale accumulators (all 8 elems of a lane share the same q)
#pragma unroll
        for (int t = 0; t < 4; ++t)
#pragma unroll
            for (int j = 0; j < 8; ++j) acc[t][j] *= alpha;

        // out^T[d,q] += V^T[d, k-block] x P^T[k-block, q]
        acc[0] = wmma16(av0, pf.v, acc[0]);
        acc[1] = wmma16(av1, pf.v, acc[1]);
        acc[2] = wmma16(av2, pf.v, acc[2]);
        acc[3] = wmma16(av3, pf.v, acc[3]);

        async_wait0();     // async fill LDS-visible before the barrier
        __syncthreads();
        cur ^= 1;
    }

    float inv = 1.0f / li;
    int qn = lane & 15;
    size_t rowbase = (((size_t)b * NS) + (size_t)qt * 16 + qn) * HID + (size_t)h * ND;
#pragma unroll
    for (int t = 0; t < 4; ++t)
#pragma unroll
        for (int j = 0; j < 8; ++j)
            AOb[rowbase + t * 16 + j + 8 * half] = (_Float16)(acc[t][j] * inv);
}

// ---------------------------------------------------------------------------
// Host launch
// ---------------------------------------------------------------------------
extern "C" void kernel_launch(void* const* d_in, const int* in_sizes, int n_in,
                              void* d_out, int out_size, void* d_ws, size_t ws_size,
                              hipStream_t stream) {
    const float* x    = (const float*)d_in[0];
    const float* mask = (const float*)d_in[1];
    const float* wq   = (const float*)d_in[2];
    const float* bq   = (const float*)d_in[3];
    const float* wk   = (const float*)d_in[4];
    const float* bk   = (const float*)d_in[5];
    const float* wv   = (const float*)d_in[6];
    const float* bv   = (const float*)d_in[7];
    const float* wo   = (const float*)d_in[8];
    const float* bo   = (const float*)d_in[9];
    float* out = (float*)d_out;

    const size_t NM = (size_t)NB * NS;         // 4096 rows
    const size_t XE = NM * HID;                // 4M elems
    const size_t WE = (size_t)HID * HID;       // 1M elems

    _Float16* ws  = (_Float16*)d_ws;
    _Float16* Xb  = ws;                        // [4096,1024]
    _Float16* Wqb = Xb  + XE;
    _Float16* Wkb = Wqb + WE;
    _Float16* Wvb = Wkb + WE;
    _Float16* Wob = Wvb + WE;
    _Float16* Qb  = Wob + WE;                  // [B,H,S,D]
    _Float16* Kb  = Qb  + XE;                  // [B,H,S,D]
    _Float16* Vtb = Kb  + XE;                  // [B,H,D,S]
    _Float16* AOb = Vtb + XE;                  // [4096,1024]

    dim3 blk(256);
    cvt_kernel<<<(int)(XE / 256), blk, 0, stream>>>(x,  Xb,  (int)XE);
    cvt_kernel<<<(int)(WE / 256), blk, 0, stream>>>(wq, Wqb, (int)WE);
    cvt_kernel<<<(int)(WE / 256), blk, 0, stream>>>(wk, Wkb, (int)WE);
    cvt_kernel<<<(int)(WE / 256), blk, 0, stream>>>(wv, Wvb, (int)WE);
    cvt_kernel<<<(int)(WE / 256), blk, 0, stream>>>(wo, Wob, (int)WE);

    // 512 blocks x 8 waves (wave32)
    gemm_nt_kernel<<<512, blk, 0, stream>>>(Xb, Wqb, bq, Qb,  nullptr, 0);
    gemm_nt_kernel<<<512, blk, 0, stream>>>(Xb, Wkb, bk, Kb,  nullptr, 0);
    gemm_nt_kernel<<<512, blk, 0, stream>>>(Xb, Wvb, bv, Vtb, nullptr, 2);

    attn_kernel<<<512, blk, 0, stream>>>(Qb, Kb, Vtb, mask, AOb);

    gemm_nt_kernel<<<512, blk, 0, stream>>>(AOb, Wob, bo, nullptr, out, 3);
}
